// EfficientViTAttention_24876450579168
// MI455X (gfx1250) — matrix operand order repaired and checked by reading, not compiled, not run on hardware
//
#include <hip/hip_runtime.h>

typedef __attribute__((ext_vector_type(16))) _Float16 v16h;
typedef __attribute__((ext_vector_type(8)))  float    v8f;

#define DIMC 128
#define HWN  16384
#define WIMG 128
#define HIMG 128
#define NB   4

__device__ __forceinline__ float hswish_f(float x) {
    float t = fminf(fmaxf(x + 3.0f, 0.0f), 6.0f);
    return x * t * (1.0f / 6.0f);
}

// ---- WMMA fragment loads from LDS (layouts per CDNA5 ISA 7.12.2) ----
// A tile in LDS: [64][32] f16, M-major. Lane l(0..15): M=m0+l, K in {lh*8.., 16+lh*8..}
__device__ __forceinline__ v16h load_a_frag(const _Float16* As, int mtile, int lane) {
    const int l15 = lane & 15, lh = lane >> 4;
    const _Float16* row = As + (mtile * 16 + l15) * 32 + (lh << 3);
    v16h a;
    ((float4*)&a)[0] = *(const float4*)(row);
    ((float4*)&a)[1] = *(const float4*)(row + 16);
    return a;
}
// B tile in LDS: [64][32] f16, N-major (B^T). Lane l(0..15): N=n0+l, K in {lh*8.., 16+lh*8..}
__device__ __forceinline__ v16h load_b_frag(const _Float16* Bs, int ntile16, int lane) {
    const int l15 = lane & 15, lh = lane >> 4;
    const _Float16* row = Bs + (ntile16 * 16 + l15) * 32 + (lh << 3);
    v16h b;
    ((float4*)&b)[0] = *(const float4*)(row);
    ((float4*)&b)[1] = *(const float4*)(row + 16);
    return b;
}

// ---- Async A-tile staging: packed f16 weights [M][K] -> LDS [64][32] ----
// Each thread issues one 16B async copy; wave covers 512B; 8 waves cover 4KB tile.
__device__ __forceinline__ void stage_a_async(_Float16* As, const _Float16* wpack,
                                              int m0, int K, int k0, int t) {
    const int ml = t >> 2;               // 0..63  (M within tile)
    const int kb = (t & 3) << 3;         // 0,8,16,24 (K within chunk)
    const _Float16* g = wpack + (size_t)(m0 + ml) * K + k0 + kb;
    const unsigned lds = (unsigned)(uintptr_t)(As + ml * 32 + kb);  // low 32b = LDS offset
    asm volatile("global_load_async_to_lds_b128 %0, %1, off"
                 :: "v"(lds), "v"((unsigned long long)(uintptr_t)g)
                 : "memory");
}
__device__ __forceinline__ void wait_async_and_sync() {
    asm volatile("s_wait_asynccnt 0x0" ::: "memory");
    __syncthreads();
}

// ---- B-tile staging: lane = pixel (fully coalesced 128B/instruction),
//      8 strided channels per thread, one ds_store_b128 per thread.
//      Branch-free edge handling: clamped address + value select.
__device__ __forceinline__ void stage_b_conv(_Float16* Bs, const float* __restrict__ x,
                                             int b, int h, int w0, int kc, int t) {
    const int rs  = kc >> 2;
    const int ci0 = (kc & 3) << 5;
    const int r = rs / 3, s = rs - 3 * r;
    const int hh = h + r - 1;
    const bool hok = (hh >= 0) && (hh < HIMG);
    const int p  = t & 63;               // pixel within tile
    const int cb = (t >> 6) << 3;        // channel sub-block: 0,8,16,24
    const int ww = w0 + s - 1 + p;
    const bool ok = hok && (ww >= 0) && (ww < WIMG);
    const int wc = ww < 0 ? 0 : (ww > WIMG - 1 ? WIMG - 1 : ww);
    const float* src = x + ((size_t)((b * DIMC + ci0 + cb) * HIMG + (hok ? hh : 0)) * WIMG) + wc;
    _Float16 vals[8];
#pragma unroll
    for (int j = 0; j < 8; ++j) {
        const float v = src[(size_t)j * HWN];     // unconditional, clamped address
        vals[j] = (_Float16)(ok ? v : 0.0f);
    }
    *(float4*)(Bs + p * 32 + cb) = *(const float4*)vals;
}

template<typename BT>
__device__ __forceinline__ void stage_b_1x1(_Float16* Bs, const BT* __restrict__ xin,
                                            int b, int hwbase, int K, int k0, int t) {
    const int p  = t & 63;
    const int cb = (t >> 6) << 3;
    const BT* src = xin + ((size_t)(b * K + k0 + cb) << 14) + hwbase + p;
    _Float16 vals[8];
#pragma unroll
    for (int j = 0; j < 8; ++j) vals[j] = (_Float16)src[(size_t)j << 14];
    *(float4*)(Bs + p * 32 + cb) = *(const float4*)vals;
}

// ---------------------------------------------------------------------------
// Weight packing (run once, tiny): f32 -> f16, conv3x3 reordered to implicit-
// GEMM K order k = (r*3+s)*128 + ci so every 32-wide K-chunk is contiguous.
// ---------------------------------------------------------------------------
__global__ void pack_w3x3_f16(const float* __restrict__ w, _Float16* __restrict__ out, int total)
{
    const int idx = blockIdx.x * blockDim.x + threadIdx.x;
    if (idx >= total) return;
    const int co = idx / 1152, k = idx % 1152;
    const int rs = k >> 7, ci = k & 127;
    const int r = rs / 3, s = rs - 3 * r;
    out[idx] = (_Float16)w[((co * DIMC + ci) * 3 + r) * 3 + s];
}
__global__ void pack_f16(const float* __restrict__ w, _Float16* __restrict__ out, int n)
{
    const int i = blockIdx.x * blockDim.x + threadIdx.x;
    if (i < n) out[i] = (_Float16)w[i];
}

// ---------------------------------------------------------------------------
// conv3x3 as implicit GEMM with WMMA; double-buffered LDS; async A staging.
// grid: (1024 N-tiles, M/64), block 256
// ---------------------------------------------------------------------------
__global__ __launch_bounds__(256)
void conv3x3_wmma(const float* __restrict__ x, const _Float16* __restrict__ wpack,
                  const float* __restrict__ bias, _Float16* __restrict__ out,
                  int Ctot, int co_off)
{
    __shared__ _Float16 As[2][64 * 32];
    __shared__ _Float16 Bs[2][64 * 32];

    const int t = threadIdx.x;
    const int lane = t & 31;
    const int wave = t >> 5;
    const int wave_m = wave & 3;
    const int wave_n = wave >> 2;

    const int nt  = blockIdx.x;
    const int b   = nt >> 8;
    const int rem = nt & 255;
    const int h   = rem >> 1;
    const int w0  = (rem & 1) << 6;
    const int m0  = blockIdx.y << 6;

    v8f acc0 = {}; v8f acc1 = {};

    stage_a_async(As[0], wpack, m0, 1152, 0, t);
    stage_b_conv(Bs[0], x, b, h, w0, 0, t);

    for (int kc = 0; kc < 36; ++kc) {
        const int cur = kc & 1;
        wait_async_and_sync();                 // stage(kc) landed in LDS
        if (kc + 1 < 36) {                     // prefetch next tile into other buffer
            stage_a_async(As[cur ^ 1], wpack, m0, 1152, (kc + 1) << 5, t);
            stage_b_conv(Bs[cur ^ 1], x, b, h, w0, kc + 1, t);
        }
        v16h af = load_a_frag(As[cur], wave_m, lane);
        v16h b0 = load_b_frag(Bs[cur], wave_n * 2 + 0, lane);
        v16h b1 = load_b_frag(Bs[cur], wave_n * 2 + 1, lane);
        acc0 = __builtin_amdgcn_wmma_f32_16x16x32_f16(false, af, false, b0, (short)0, acc0, false, false);
        acc1 = __builtin_amdgcn_wmma_f32_16x16x32_f16(false, af, false, b1, (short)0, acc1, false, false);
    }

    const int l15 = lane & 15, lh = lane >> 4;
    const int ncol = w0 + (wave_n << 5) + l15;
#pragma unroll
    for (int rr = 0; rr < 8; ++rr) {
        const int co = m0 + (wave_m << 4) + (lh << 3) + rr;
        const float bv = bias[co];
        const int base = ((b * Ctot + co_off + co) << 14) + h * WIMG;
        out[base + ncol]      = (_Float16)(acc0[rr] + bv);
        out[base + ncol + 16] = (_Float16)(acc1[rr] + bv);
    }
}

// ---------------------------------------------------------------------------
// 1x1 conv GEMM. MODE 1: BN + residual -> f32 out. MODE 2: bias + hswish -> f16.
// Double-buffered LDS; async A staging from packed f16 weights.
// ---------------------------------------------------------------------------
template<int MODE, typename BT>
__global__ __launch_bounds__(256)
void gemm1x1_wmma(const BT* __restrict__ xin, const _Float16* __restrict__ wpack,
                  const float* __restrict__ bias,
                  const float* __restrict__ bng, const float* __restrict__ bnb,
                  const float* __restrict__ bnm, const float* __restrict__ bnv,
                  const float* __restrict__ resid,
                  float* __restrict__ outf, _Float16* __restrict__ outh,
                  int M, int K)
{
    __shared__ _Float16 As[2][64 * 32];
    __shared__ _Float16 Bs[2][64 * 32];

    const int t = threadIdx.x;
    const int lane = t & 31;
    const int wave = t >> 5;
    const int wave_m = wave & 3;
    const int wave_n = wave >> 2;

    const int nt  = blockIdx.x;
    const int b   = nt >> 8;
    const int rem = nt & 255;
    const int h   = rem >> 1;
    const int w0  = (rem & 1) << 6;
    const int m0  = blockIdx.y << 6;
    const int hwbase = h * WIMG + w0;

    v8f acc0 = {}; v8f acc1 = {};

    const int nkc = K >> 5;
    stage_a_async(As[0], wpack, m0, K, 0, t);
    stage_b_1x1<BT>(Bs[0], xin, b, hwbase, K, 0, t);

    for (int kc = 0; kc < nkc; ++kc) {
        const int cur = kc & 1;
        wait_async_and_sync();
        if (kc + 1 < nkc) {
            stage_a_async(As[cur ^ 1], wpack, m0, K, (kc + 1) << 5, t);
            stage_b_1x1<BT>(Bs[cur ^ 1], xin, b, hwbase, K, (kc + 1) << 5, t);
        }
        v16h af = load_a_frag(As[cur], wave_m, lane);
        v16h b0 = load_b_frag(Bs[cur], wave_n * 2 + 0, lane);
        v16h b1 = load_b_frag(Bs[cur], wave_n * 2 + 1, lane);
        acc0 = __builtin_amdgcn_wmma_f32_16x16x32_f16(false, af, false, b0, (short)0, acc0, false, false);
        acc1 = __builtin_amdgcn_wmma_f32_16x16x32_f16(false, af, false, b1, (short)0, acc1, false, false);
    }

    const int l15 = lane & 15, lh = lane >> 4;
    const int ncol = w0 + (wave_n << 5) + l15;
#pragma unroll
    for (int rr = 0; rr < 8; ++rr) {
        const int co  = m0 + (wave_m << 4) + (lh << 3) + rr;
        const int idx = ((b * M + co) << 14) + h * WIMG + ncol;
        float v0 = acc0[rr], v1 = acc1[rr];
        if (MODE == 2) {
            const float bv = bias[co];
            outh[idx]      = (_Float16)hswish_f(v0 + bv);
            outh[idx + 16] = (_Float16)hswish_f(v1 + bv);
        } else { // MODE 1: BN + residual, f32 out
            const float sc = bng[co] * rsqrtf(bnv[co] + 1e-5f);
            const float sh = bnb[co] - bnm[co] * sc;
            outf[idx]      = v0 * sc + sh + resid[idx];
            outf[idx + 16] = v1 * sc + sh + resid[idx + 16];
        }
    }
}

// ---------------------------------------------------------------------------
// Depthwise conv (5x5 or 3x3), optional bias + hswish. One thread per output.
// ---------------------------------------------------------------------------
__global__ void dw_conv_kernel(const _Float16* __restrict__ in, const float* __restrict__ wgt,
                               const float* __restrict__ bias, _Float16* __restrict__ out,
                               int C, int KS, int pad, int act, int total)
{
    const int idx = blockIdx.x * blockDim.x + threadIdx.x;
    if (idx >= total) return;
    const int n = idx & (HWN - 1);
    const int c = (idx >> 14) % C;
    const int b = idx / (C << 14);
    const int h = n >> 7, w = n & 127;
    const _Float16* src = in + ((size_t)(b * C + c) << 14);
    const float* wp = wgt + c * KS * KS;
    float s = 0.0f;
    for (int r = 0; r < KS; ++r) {
        const int hh = h + r - pad;
        if (hh < 0 || hh >= HIMG) continue;
        for (int q = 0; q < KS; ++q) {
            const int ww = w + q - pad;
            if (ww < 0 || ww >= WIMG) continue;
            s += (float)src[(hh << 7) + ww] * wp[r * KS + q];
        }
    }
    if (bias) s += bias[c];
    if (act) s = hswish_f(s);
    out[idx] = (_Float16)s;
}

// Pointwise grouped conv (384 out, 8 in per group, 48 groups), no bias
__global__ void pw_group8(const _Float16* __restrict__ in, const float* __restrict__ wgt,
                          _Float16* __restrict__ out, int total)
{
    const int idx = blockIdx.x * blockDim.x + threadIdx.x;
    if (idx >= total) return;
    const int n = idx & (HWN - 1);
    const int c = (idx >> 14) % 384;
    const int b = idx / (384 << 14);
    const int g = c >> 3;
    const _Float16* src = in + ((size_t)(b * 384 + g * 8) << 14) + n;
    const float* wr = wgt + c * 8;
    float s = 0.0f;
#pragma unroll
    for (int j = 0; j < 8; ++j) s += (float)src[(size_t)j << 14] * wr[j];
    out[idx] = (_Float16)s;
}

// multi = concat(qkv[0..383], ms[384..767]) channel lookup
__device__ __forceinline__ float load_multi(const _Float16* qkv, const _Float16* ms,
                                            int b, int c, int n)
{
    return (c < 384) ? (float)qkv[((size_t)(b * 384 + c) << 14) + n]
                     : (float)ms [((size_t)(b * 384 + c - 384) << 14) + n];
}

// Phase 1: kv[b,g,9,8] = sum_n v_d(n) * relu(k_e(n)); grid = B*32*8 chunks
__global__ __launch_bounds__(256)
void att_kv(const _Float16* __restrict__ qkv, const _Float16* __restrict__ ms,
            float* __restrict__ kvout)
{
    __shared__ float sh[72];
    const int t = threadIdx.x;
    const int blk = blockIdx.x;
    const int ch = blk & 7;
    const int g  = (blk >> 3) & 31;
    const int b  = blk >> 8;
    if (t < 72) sh[t] = 0.0f;
    __syncthreads();

    float kvl[72];
#pragma unroll
    for (int i = 0; i < 72; ++i) kvl[i] = 0.0f;

    const int n0 = (ch << 11) + (t << 3);
    for (int p = 0; p < 8; ++p) {
        const int n = n0 + p;
        float kk[8], vv[9];
#pragma unroll
        for (int e = 0; e < 8; ++e) kk[e] = fmaxf(load_multi(qkv, ms, b, 24 * g + 8 + e, n), 0.0f);
#pragma unroll
        for (int d = 0; d < 8; ++d) vv[d] = load_multi(qkv, ms, b, 24 * g + 16 + d, n);
        vv[8] = 1.0f;
#pragma unroll
        for (int d = 0; d < 9; ++d)
#pragma unroll
            for (int e = 0; e < 8; ++e) kvl[d * 8 + e] += vv[d] * kk[e];
    }
#pragma unroll
    for (int i = 0; i < 72; ++i) {
        float v = kvl[i];
        for (int off = 16; off > 0; off >>= 1) v += __shfl_xor(v, off, 32);
        if ((t & 31) == 0) atomicAdd(&sh[i], v);
    }
    __syncthreads();
    if (t < 72) atomicAdd(&kvout[(b * 32 + g) * 72 + t], sh[t]);
}

// Phase 2: out[d,n] = (kv[d,:]·q)/(kv[8,:]·q + eps); grid = B*32*64
__global__ __launch_bounds__(256)
void att_apply(const _Float16* __restrict__ qkv, const _Float16* __restrict__ ms,
               const float* __restrict__ kv, _Float16* __restrict__ att)
{
    __shared__ float sh[72];
    const int blk = blockIdx.x;
    const int chunk = blk & 63;
    const int g = (blk >> 6) & 31;
    const int b = blk >> 11;
    if (threadIdx.x < 72) sh[threadIdx.x] = kv[(b * 32 + g) * 72 + threadIdx.x];
    __syncthreads();
    const int n = (chunk << 8) + threadIdx.x;
    float q[8];
#pragma unroll
    for (int e = 0; e < 8; ++e) q[e] = fmaxf(load_multi(qkv, ms, b, 24 * g + e, n), 0.0f);
    float den = 1e-15f;
#pragma unroll
    for (int e = 0; e < 8; ++e) den += sh[64 + e] * q[e];
    const float rden = 1.0f / den;
#pragma unroll
    for (int d = 0; d < 8; ++d) {
        float num = 0.0f;
#pragma unroll
        for (int e = 0; e < 8; ++e) num += sh[d * 8 + e] * q[e];
        att[((size_t)(b * 256 + g * 8 + d) << 14) + n] = (_Float16)(num * rden);
    }
}

__global__ void zero_f32(float* __restrict__ p, int n)
{
    const int i = blockIdx.x * blockDim.x + threadIdx.x;
    if (i < n) p[i] = 0.0f;
}

// ---------------------------------------------------------------------------
extern "C" void kernel_launch(void* const* d_in, const int* in_sizes, int n_in,
                              void* d_out, int out_size, void* d_ws, size_t ws_size,
                              hipStream_t stream)
{
    (void)in_sizes; (void)n_in; (void)out_size; (void)ws_size;
    const float* ref   = (const float*)d_in[0];
    const float* other = (const float*)d_in[1];
    const float* wq = (const float*)d_in[2];   const float* bq = (const float*)d_in[3];
    const float* wk = (const float*)d_in[4];   const float* bk = (const float*)d_in[5];
    const float* wv = (const float*)d_in[6];   const float* bv = (const float*)d_in[7];
    const float* agg_dw = (const float*)d_in[8];
    const float* agg_pw = (const float*)d_in[9];
    const float* projw  = (const float*)d_in[10];
    const float* bn1g = (const float*)d_in[11]; const float* bn1b = (const float*)d_in[12];
    const float* bn1m = (const float*)d_in[13]; const float* bn1v = (const float*)d_in[14];
    const float* mb1w = (const float*)d_in[15]; const float* mb1b = (const float*)d_in[16];
    const float* mb2w = (const float*)d_in[17]; const float* mb2b = (const float*)d_in[18];
    const float* mb3w = (const float*)d_in[19];
    const float* bn2g = (const float*)d_in[20]; const float* bn2b = (const float*)d_in[21];
    const float* bn2m = (const float*)d_in[22]; const float* bn2v = (const float*)d_in[23];
    float* out = (float*)d_out;

    char* ws = (char*)d_ws;
    const size_t QKV_B = (size_t)NB * 384 * HWN * 2;                  // 50,331,648
    _Float16* qkv = (_Float16*)(ws);
    _Float16* tmp = (_Float16*)(ws + QKV_B);
    _Float16* msb = (_Float16*)(ws + 2 * QKV_B);
    _Float16* att = (_Float16*)(ws + 3 * QKV_B);
    float* attended = (float*)(ws + 3 * QKV_B + (size_t)NB * 256 * HWN * 2);
    float* kv = (float*)((char*)attended + (size_t)NB * 128 * HWN * 4);
    _Float16* h1 = qkv;                                               // reuse qkv+tmp (dead)
    _Float16* h2 = (_Float16*)((char*)kv + 65536);
    // packed f16 weights after h2 (~1.3 MB total)
    _Float16* wpq    = (_Float16*)((char*)h2 + (size_t)NB * 768 * HWN * 2);
    _Float16* wpk    = wpq + 128 * 1152;
    _Float16* wpv    = wpk + 128 * 1152;
    _Float16* wpproj = wpv + 128 * 1152;
    _Float16* wpmb1  = wpproj + 128 * 256;
    _Float16* wpmb3  = wpmb1 + 768 * 128;

    const dim3 blk(256);

    // one-time weight packing + kv zero
    pack_w3x3_f16<<<(128 * 1152 + 255) / 256, blk, 0, stream>>>(wq, wpq, 128 * 1152);
    pack_w3x3_f16<<<(128 * 1152 + 255) / 256, blk, 0, stream>>>(wk, wpk, 128 * 1152);
    pack_w3x3_f16<<<(128 * 1152 + 255) / 256, blk, 0, stream>>>(wv, wpv, 128 * 1152);
    pack_f16<<<(128 * 256 + 255) / 256, blk, 0, stream>>>(projw, wpproj, 128 * 256);
    pack_f16<<<(768 * 128 + 255) / 256, blk, 0, stream>>>(mb1w, wpmb1, 768 * 128);
    pack_f16<<<(128 * 768 + 255) / 256, blk, 0, stream>>>(mb3w, wpmb3, 128 * 768);
    zero_f32<<<dim3((NB * 32 * 72 + 255) / 256), blk, 0, stream>>>(kv, NB * 32 * 72);

    const dim3 g3(1024, 2);  // 1024 N-tiles x (128/64) M-tiles
    conv3x3_wmma<<<g3, blk, 0, stream>>>(ref,   wpq, bq, qkv, 384, 0);
    conv3x3_wmma<<<g3, blk, 0, stream>>>(other, wpk, bk, qkv, 384, 128);
    conv3x3_wmma<<<g3, blk, 0, stream>>>(other, wpv, bv, qkv, 384, 256);

    {
        const int total = NB * 384 * HWN;
        dw_conv_kernel<<<(total + 255) / 256, blk, 0, stream>>>(qkv, agg_dw, nullptr, tmp, 384, 5, 2, 0, total);
        pw_group8<<<(total + 255) / 256, blk, 0, stream>>>(tmp, agg_pw, msb, total);
    }

    att_kv   <<<dim3(NB * 32 * 8),  blk, 0, stream>>>(qkv, msb, kv);
    att_apply<<<dim3(NB * 32 * 64), blk, 0, stream>>>(qkv, msb, kv, att);

    // attn_proj (K=256) + BN1 + residual(ref) -> attended (f32)
    gemm1x1_wmma<1, _Float16><<<dim3(1024, 2), blk, 0, stream>>>(
        att, wpproj, nullptr, bn1g, bn1b, bn1m, bn1v, ref, attended, nullptr, 128, 256);

    // mb1 (M=768, K=128) + bias + hswish -> h1 (f16)
    gemm1x1_wmma<2, float><<<dim3(1024, 12), blk, 0, stream>>>(
        attended, wpmb1, mb1b, nullptr, nullptr, nullptr, nullptr, nullptr, nullptr, h1, 768, 128);

    // mb2 depthwise 3x3 + bias + hswish -> h2 (f16)
    {
        const int total = NB * 768 * HWN;
        dw_conv_kernel<<<(total + 255) / 256, blk, 0, stream>>>(h1, mb2w, mb2b, h2, 768, 3, 1, 1, total);
    }

    // mb3 (K=768) + BN2 + residual(attended) -> d_out (f32)
    gemm1x1_wmma<1, _Float16><<<dim3(1024, 2), blk, 0, stream>>>(
        h2, wpmb3, nullptr, bn2g, bn2b, bn2m, bn2v, attended, out, nullptr, 128, 768);
}
